// HypAgg_75187697484266
// MI455X (gfx1250) — compile-verified, matrix operand order checked by reading, not run on hardware
//
#include <hip/hip_runtime.h>
#include <hip/hip_bf16.h>

typedef _Float16 half_t;
typedef __attribute__((ext_vector_type(16))) _Float16 v16h;
typedef __attribute__((ext_vector_type(8)))  _Float16 v8h;
typedef __attribute__((ext_vector_type(8)))  float    v8f;

#define WB() __builtin_amdgcn_wave_barrier()

__device__ inline float fast_tanh(float z) {
#if __has_builtin(__builtin_amdgcn_tanhf)
    return __builtin_amdgcn_tanhf(z);
#else
    // branch-free: tanh(z) = 2*sigmoid(2z) - 1
    return 2.0f / (1.0f + __expf(-2.0f * z)) - 1.0f;
#endif
}

__device__ inline float gelu_f(float x) {
    float z = 0.7978845608028654f * (x + 0.044715f * x * x * x);
    return 0.5f * x * (1.0f + fast_tanh(z));
}

// monotone float -> uint key for atomicMax over floats
__device__ inline unsigned fkey(float f) {
    unsigned u = __float_as_uint(f);
    return (u & 0x80000000u) ? ~u : (u | 0x80000000u);
}
__device__ inline float funkey(unsigned k) {
    unsigned u = (k & 0x80000000u) ? (k ^ 0x80000000u) : ~k;
    return __uint_as_float(u);
}

// ---------------- kernel 1: logmap0 (c=1) -> f16 rows ----------------
__global__ __launch_bounds__(256) void k_logmap(const float* __restrict__ x,
                                                half_t* __restrict__ x0h, int n) {
    int wid  = (int)((blockIdx.x * blockDim.x + threadIdx.x) >> 5);
    int lane = threadIdx.x & 31;
    if (wid >= n) return;
    const float* row = x + (size_t)wid * 64;
    float a0 = row[lane], a1 = row[lane + 32];
    float ss = a0 * a0 + a1 * a1;
#pragma unroll
    for (int o = 16; o > 0; o >>= 1) ss += __shfl_xor(ss, o, 32);
    float nrm = fmaxf(sqrtf(ss), 1e-7f);
    float arg = fminf(nrm, 1.0f - 1e-6f);
    float sc  = atanhf(arg) / nrm;
    x0h[(size_t)wid * 64 + lane]      = (half_t)(a0 * sc);
    x0h[(size_t)wid * 64 + lane + 32] = (half_t)(a1 * sc);
}

// ---------- kernel 2: pack f32 weights into WMMA-B fragment layout ----------
// dst[frag*512 + lane*16 + i] = W[k, col]   with
//   frag = j*NB + nb ; kb = (lane>>4)*8 ; k = 32j + kb + (i<8 ? i : i+8)
//   col  = nb*16 + (lane&15)
__global__ __launch_bounds__(256) void k_packw(const float* __restrict__ w,
                                               half_t* __restrict__ dst,
                                               int K, int Ncol) {
    int idx = blockIdx.x * blockDim.x + threadIdx.x;
    if (idx >= K * Ncol) return;
    int i = idx & 15, lane = (idx >> 4) & 31, f = idx >> 9;
    int NB = Ncol >> 4;
    int j = f / NB, nb = f % NB;
    int kb  = (lane >> 4) * 8;
    int k   = 32 * j + kb + (i < 8 ? i : i + 8);
    int col = nb * 16 + (lane & 15);
    dst[idx] = (half_t)w[k * Ncol + col];
}

// ---------------- kernel 3: init accumulators ----------------
__global__ __launch_bounds__(256) void k_init(float* __restrict__ zbase, size_t nz,
                                              unsigned* __restrict__ maxk, size_t nm) {
    size_t i = (size_t)blockIdx.x * blockDim.x + threadIdx.x;
    if (i < nz) zbase[i] = 0.0f;
    if (i < nm) maxk[i] = 0x007FFFFFu;  // fkey(-inf)
}

// build one 16x32 f16 A fragment from an LDS row (contiguous 8-half runs)
__device__ inline v16h frag_from_row(const half_t* row, int j, int kb) {
    v8h p0 = *(const v8h*)(row + 32 * j + kb);
    v8h p1 = *(const v8h*)(row + 32 * j + kb + 16);
    return __builtin_shufflevector(p0, p1, 0, 1, 2, 3, 4, 5, 6, 7,
                                   8, 9, 10, 11, 12, 13, 14, 15);
}
// edge-input fragment: row = [xr(0..63) | xs(64..127)], features 128..191 = xs-xr
__device__ inline v16h edge_frag(const half_t* row, int j, int kb) {
    v8h part[2];
#pragma unroll
    for (int s2 = 0; s2 < 2; s2++) {
        int k0 = 32 * j + kb + 16 * s2;
        v8h val;
        if (k0 < 128) {
            val = *(const v8h*)(row + k0);
        } else {
            v8h xr = *(const v8h*)(row + (k0 - 128));
            v8h xs = *(const v8h*)(row + (k0 - 64));
            val = xs - xr;
        }
        part[s2] = val;
    }
    return __builtin_shufflevector(part[0], part[1], 0, 1, 2, 3, 4, 5, 6, 7,
                                   8, 9, 10, 11, 12, 13, 14, 15);
}
__device__ inline v16h bfrag(const half_t* pw, int fragIdx, int lane) {
    return *(const v16h*)(pw + (size_t)fragIdx * 512 + lane * 16);
}

// ---------------- kernel 4: edge MLP + scatter (pass A), M=32 per wave --------
__global__ __launch_bounds__(128) void k_edge(
    const half_t* __restrict__ x0h, const int* __restrict__ adj,
    const half_t* __restrict__ pw0, const float* __restrict__ eb0,
    const half_t* __restrict__ pw1, const float* __restrict__ eb1,
    const half_t* __restrict__ pw2, const float* __restrict__ eb2,
    half_t* __restrict__ hbuf,
    float* __restrict__ gsum, float* __restrict__ gsum2,
    unsigned* __restrict__ gmax, int E, int ntiles) {
    __shared__ __align__(16) half_t sIn[4][32 * 128];  // per wave: 32 x [xr|xs]
    __shared__ int sR[4][32];

    int w = threadIdx.x >> 5, lane = threadIdx.x & 31;
    int tile = blockIdx.x * 4 + w;
    if (tile >= ntiles) return;
    int e0 = tile * 32;

    __builtin_prefetch(pw0, 0, 0);
    __builtin_prefetch(pw1, 0, 0);
    __builtin_prefetch(pw2, 0, 0);

    // stage xr (feat 0..63) and xs (64..127) for 32 edges; one edge per lane
    {
        int e = e0 + lane; if (e >= E) e = E - 1;
        int rn = adj[E + e];   // receiver
        int sn = adj[e];       // sender
        sR[w][lane] = rn;
        const uint4* srcr = (const uint4*)(x0h + (size_t)rn * 64);
        const uint4* srcs = (const uint4*)(x0h + (size_t)sn * 64);
        uint4* dst = (uint4*)&sIn[w][lane * 128];
#pragma unroll
        for (int q = 0; q < 8; q++) { dst[q] = srcr[q]; dst[8 + q] = srcs[q]; }
    }
    WB();

    int m = lane & 15, hb = lane >> 4, kb = hb * 8;
    int rvA[8], rvB[8];
#pragma unroll
    for (int v = 0; v < 8; v++) {
        rvA[v] = sR[w][v + 8 * hb];
        rvB[v] = sR[w][16 + v + 8 * hb];
    }
    const half_t* rowA = &sIn[w][m * 128];
    const half_t* rowB = &sIn[w][(16 + m) * 128];

    // ---- layer 0: [32,192] x [192,128] ----
    v16h A0a[6], A0b[6];
#pragma unroll
    for (int j = 0; j < 6; j++) {
        A0a[j] = edge_frag(rowA, j, kb);
        A0b[j] = edge_frag(rowB, j, kb);
    }
    WB();
    half_t* act = &sIn[w][0];  // reuse staging region: 32x128 activations
#pragma unroll
    for (int nb = 0; nb < 8; nb++) {
        v8f a0 = {}, a1 = {};
#pragma unroll
        for (int j = 0; j < 6; j++) {
            v16h B = bfrag(pw0, j * 8 + nb, lane);
            a0 = __builtin_amdgcn_wmma_f32_16x16x32_f16(false, A0a[j], false, B,
                                                        (short)0, a0, false, false);
            a1 = __builtin_amdgcn_wmma_f32_16x16x32_f16(false, A0b[j], false, B,
                                                        (short)0, a1, false, false);
        }
        float bias = eb0[nb * 16 + m];
#pragma unroll
        for (int v = 0; v < 8; v++) {
            act[(v + 8 * hb) * 128 + nb * 16 + m]        = (half_t)gelu_f(a0[v] + bias);
            act[(16 + v + 8 * hb) * 128 + nb * 16 + m]   = (half_t)gelu_f(a1[v] + bias);
        }
    }
    WB();

    // ---- layer 1: [32,128] x [128,128] ----
    v16h A1a[4], A1b[4];
#pragma unroll
    for (int j = 0; j < 4; j++) {
        A1a[j] = frag_from_row(&act[m * 128], j, kb);
        A1b[j] = frag_from_row(&act[(16 + m) * 128], j, kb);
    }
    WB();
#pragma unroll
    for (int nb = 0; nb < 8; nb++) {
        v8f a0 = {}, a1 = {};
#pragma unroll
        for (int j = 0; j < 4; j++) {
            v16h B = bfrag(pw1, j * 8 + nb, lane);
            a0 = __builtin_amdgcn_wmma_f32_16x16x32_f16(false, A1a[j], false, B,
                                                        (short)0, a0, false, false);
            a1 = __builtin_amdgcn_wmma_f32_16x16x32_f16(false, A1b[j], false, B,
                                                        (short)0, a1, false, false);
        }
        float bias = eb1[nb * 16 + m];
#pragma unroll
        for (int v = 0; v < 8; v++) {
            act[(v + 8 * hb) * 128 + nb * 16 + m]        = (half_t)gelu_f(a0[v] + bias);
            act[(16 + v + 8 * hb) * 128 + nb * 16 + m]   = (half_t)gelu_f(a1[v] + bias);
        }
    }
    WB();

    // ---- layer 2: [32,128] x [128,64] + scatter ----
    v16h A2a[4], A2b[4];
#pragma unroll
    for (int j = 0; j < 4; j++) {
        A2a[j] = frag_from_row(&act[m * 128], j, kb);
        A2b[j] = frag_from_row(&act[(16 + m) * 128], j, kb);
    }
#pragma unroll
    for (int nb = 0; nb < 4; nb++) {
        v8f a0 = {}, a1 = {};
#pragma unroll
        for (int j = 0; j < 4; j++) {
            v16h B = bfrag(pw2, j * 4 + nb, lane);
            a0 = __builtin_amdgcn_wmma_f32_16x16x32_f16(false, A2a[j], false, B,
                                                        (short)0, a0, false, false);
            a1 = __builtin_amdgcn_wmma_f32_16x16x32_f16(false, A2b[j], false, B,
                                                        (short)0, a1, false, false);
        }
        float bias = eb2[nb * 16 + m];
        int feat = nb * 16 + m;
#pragma unroll
        for (int v = 0; v < 8; v++) {
            int eA = e0 + v + 8 * hb;
            if (eA < E) {
                float hv = a0[v] + bias;
                size_t off = (size_t)rvA[v] * 64 + feat;
                atomicAdd(&gsum[off], hv);
                atomicAdd(&gsum2[off], hv * hv);
                atomicMax(&gmax[off], fkey(hv));
                hbuf[(size_t)eA * 64 + feat] = (half_t)hv;
            }
            int eB = e0 + 16 + v + 8 * hb;
            if (eB < E) {
                float hv = a1[v] + bias;
                size_t off = (size_t)rvB[v] * 64 + feat;
                atomicAdd(&gsum[off], hv);
                atomicAdd(&gsum2[off], hv * hv);
                atomicMax(&gmax[off], fkey(hv));
                hbuf[(size_t)eB * 64 + feat] = (half_t)hv;
            }
        }
    }
}

// ---------------- kernel 5: softmax accumulation (pass B) ----------------
__global__ __launch_bounds__(256) void k_soft(
    const half_t* __restrict__ hbuf, const int* __restrict__ adj,
    const unsigned* __restrict__ gmax,
    float* __restrict__ gs1, float* __restrict__ gw1,
    float* __restrict__ gs10, float* __restrict__ gw10,
    float* __restrict__ gcnt, int E) {
    int wid  = (int)((blockIdx.x * blockDim.x + threadIdx.x) >> 5);
    int lane = threadIdx.x & 31;
    if (wid >= E) return;
    int rn = adj[E + wid];
    if (lane == 0) atomicAdd(&gcnt[rn], 1.0f);
#pragma unroll
    for (int t = 0; t < 2; t++) {
        int feat = lane + 32 * t;
        float hv = (float)hbuf[(size_t)wid * 64 + feat];
        size_t off = (size_t)rn * 64 + feat;
        float mx = funkey(gmax[off]);  // finite: this segment has >=1 edge
        float d = hv - mx;
        float e1 = __expf(d);
        float e10 = __expf(10.0f * d);
        atomicAdd(&gs1[off], e1);
        atomicAdd(&gw1[off], hv * e1);
        atomicAdd(&gs10[off], e10);
        atomicAdd(&gw10[off], hv * e10);
    }
}

// ---------------- kernel 6: node MLP + expmap0 + proj (pass C) ----------------
__global__ __launch_bounds__(128) void k_node(
    const float* __restrict__ gsum, const float* __restrict__ gsum2,
    const float* __restrict__ gs1, const float* __restrict__ gw1,
    const float* __restrict__ gs10, const float* __restrict__ gw10,
    const float* __restrict__ gcnt,
    const half_t* __restrict__ pw0, const float* __restrict__ ab0,
    const half_t* __restrict__ pw1, const float* __restrict__ ab1,
    const half_t* __restrict__ pw2, const float* __restrict__ ab2,
    float* __restrict__ out, int N, int ntiles) {
    __shared__ __align__(16) half_t sAgg[4][16 * 256];
    __shared__ __align__(16) half_t sAct[4][16 * 128];
    __shared__ __align__(16) float  sU[4][16 * 64];
    __shared__ float sSc[4][16];

    int w = threadIdx.x >> 5, lane = threadIdx.x & 31;
    int tile = blockIdx.x * 4 + w;
    if (tile >= ntiles) return;

    __builtin_prefetch(pw0, 0, 0);
    __builtin_prefetch(pw1, 0, 0);
    __builtin_prefetch(pw2, 0, 0);

    int m = lane & 15, hb = lane >> 4, kb = hb * 8;
    int node = tile * 16 + m;
    if (node >= N) node = N - 1;

    // assemble x_agg = [mean | var | sm1 | sm10]  (f16)
    if (hb == 0) {
        float inv = 1.0f / fmaxf(gcnt[node], 1.0f);
#pragma unroll 4
        for (int j = 0; j < 64; j++) {
            size_t off = (size_t)node * 64 + j;
            float mean = gsum[off] * inv;
            float var = fmaxf(gsum2[off] * inv - mean * mean, 0.0f);
            sAgg[w][m * 256 + j] = (half_t)mean;
            sAgg[w][m * 256 + 64 + j] = (half_t)var;
        }
    } else {
#pragma unroll 4
        for (int j = 0; j < 64; j++) {
            size_t off = (size_t)node * 64 + j;
            float s1 = fmaxf(gs1[off], 1e-7f);
            float s10 = fmaxf(gs10[off], 1e-7f);
            sAgg[w][m * 256 + 128 + j] = (half_t)(gw1[off] / s1);
            sAgg[w][m * 256 + 192 + j] = (half_t)(gw10[off] / s10);
        }
    }
    WB();

    // ---- layer 0: [16,256] x [256,128] ----
    v16h A0[8];
#pragma unroll
    for (int j = 0; j < 8; j++) A0[j] = frag_from_row(&sAgg[w][m * 256], j, kb);
    WB();
#pragma unroll
    for (int nb = 0; nb < 8; nb++) {
        v8f acc = {};
#pragma unroll
        for (int j = 0; j < 8; j++)
            acc = __builtin_amdgcn_wmma_f32_16x16x32_f16(false, A0[j], false,
                                                         bfrag(pw0, j * 8 + nb, lane),
                                                         (short)0, acc, false, false);
        float bias = ab0[nb * 16 + m];
#pragma unroll
        for (int v = 0; v < 8; v++)
            sAct[w][(v + 8 * hb) * 128 + nb * 16 + m] = (half_t)gelu_f(acc[v] + bias);
    }
    WB();

    // ---- layer 1: [16,128] x [128,128] ----
    v16h A1[4];
#pragma unroll
    for (int j = 0; j < 4; j++) A1[j] = frag_from_row(&sAct[w][m * 128], j, kb);
    WB();
#pragma unroll
    for (int nb = 0; nb < 8; nb++) {
        v8f acc = {};
#pragma unroll
        for (int j = 0; j < 4; j++)
            acc = __builtin_amdgcn_wmma_f32_16x16x32_f16(false, A1[j], false,
                                                         bfrag(pw1, j * 8 + nb, lane),
                                                         (short)0, acc, false, false);
        float bias = ab1[nb * 16 + m];
#pragma unroll
        for (int v = 0; v < 8; v++)
            sAct[w][(v + 8 * hb) * 128 + nb * 16 + m] = (half_t)gelu_f(acc[v] + bias);
    }
    WB();

    // ---- layer 2: [16,128] x [128,64] -> u (f32) ----
    v16h A2[4];
#pragma unroll
    for (int j = 0; j < 4; j++) A2[j] = frag_from_row(&sAct[w][m * 128], j, kb);
#pragma unroll
    for (int nb = 0; nb < 4; nb++) {
        v8f acc = {};
#pragma unroll
        for (int j = 0; j < 4; j++)
            acc = __builtin_amdgcn_wmma_f32_16x16x32_f16(false, A2[j], false,
                                                         bfrag(pw2, j * 4 + nb, lane),
                                                         (short)0, acc, false, false);
        float bias = ab2[nb * 16 + m];
#pragma unroll
        for (int v = 0; v < 8; v++)
            sU[w][(v + 8 * hb) * 64 + nb * 16 + m] = acc[v] + bias;
    }
    WB();

    // expmap0 (c=1) + proj per row
    if (lane < 16) {
        float ss = 0.0f;
#pragma unroll 8
        for (int j = 0; j < 64; j++) { float u = sU[w][lane * 64 + j]; ss += u * u; }
        float nrm = fmaxf(sqrtf(ss), 1e-7f);
        float th = fast_tanh(nrm);
        float sc = th / nrm;
        const float maxnorm = 1.0f - 1e-5f;
        float ynorm = fmaxf(th, 1e-7f);
        if (ynorm > maxnorm) sc *= maxnorm / ynorm;
        sSc[w][lane] = sc;
    }
    WB();
#pragma unroll
    for (int i = 0; i < 32; i++) {
        int idx = lane + 32 * i;
        int row = idx >> 6, col = idx & 63;
        int gn = tile * 16 + row;
        if (gn < N) out[(size_t)gn * 64 + col] = sU[w][idx] * sSc[w][row];
    }
}

// ---------------- host launcher ----------------
extern "C" void kernel_launch(void* const* d_in, const int* in_sizes, int n_in,
                              void* d_out, int out_size, void* d_ws, size_t ws_size,
                              hipStream_t stream) {
    const float* x   = (const float*)d_in[0];
    const int*   adj = (const int*)d_in[1];
    const float* ew0 = (const float*)d_in[2];  const float* eb0 = (const float*)d_in[3];
    const float* ew1 = (const float*)d_in[4];  const float* eb1 = (const float*)d_in[5];
    const float* ew2 = (const float*)d_in[6];  const float* eb2 = (const float*)d_in[7];
    const float* aw0 = (const float*)d_in[8];  const float* ab0 = (const float*)d_in[9];
    const float* aw1 = (const float*)d_in[10]; const float* ab1 = (const float*)d_in[11];
    const float* aw2 = (const float*)d_in[12]; const float* ab2 = (const float*)d_in[13];

    const int N = in_sizes[0] / 64;
    const int E = in_sizes[1] / 2;
    const int D = 64;

    // carve workspace
    char* ws = (char*)d_ws;
    auto carve = [&](size_t bytes) -> char* {
        char* p = ws;
        ws += (bytes + 255) & ~(size_t)255;
        return p;
    };
    half_t* x0h  = (half_t*)carve((size_t)N * D * sizeof(half_t));
    half_t* pe0  = (half_t*)carve(192 * 128 * sizeof(half_t));
    half_t* pe1  = (half_t*)carve(128 * 128 * sizeof(half_t));
    half_t* pe2  = (half_t*)carve(128 * 64 * sizeof(half_t));
    half_t* pa0  = (half_t*)carve(256 * 128 * sizeof(half_t));
    half_t* pa1  = (half_t*)carve(128 * 128 * sizeof(half_t));
    half_t* pa2  = (half_t*)carve(128 * 64 * sizeof(half_t));
    half_t* hbuf = (half_t*)carve((size_t)E * D * sizeof(half_t));
    unsigned* gmax = (unsigned*)carve((size_t)N * D * sizeof(unsigned));
    size_t nz = (size_t)6 * N * D + N;  // sum,sum2,s1,w1,s10,w10,cnt
    float* accum = (float*)carve(nz * sizeof(float));
    float* gsum  = accum;
    float* gsum2 = accum + (size_t)1 * N * D;
    float* gs1   = accum + (size_t)2 * N * D;
    float* gw1   = accum + (size_t)3 * N * D;
    float* gs10  = accum + (size_t)4 * N * D;
    float* gw10  = accum + (size_t)5 * N * D;
    float* gcnt  = accum + (size_t)6 * N * D;

    // 1) logmap0
    k_logmap<<<(N + 7) / 8, 256, 0, stream>>>(x, x0h, N);

    // 2) pack weights
    k_packw<<<(192 * 128 + 255) / 256, 256, 0, stream>>>(ew0, pe0, 192, 128);
    k_packw<<<(128 * 128 + 255) / 256, 256, 0, stream>>>(ew1, pe1, 128, 128);
    k_packw<<<(128 * 64 + 255) / 256, 256, 0, stream>>>(ew2, pe2, 128, 64);
    k_packw<<<(256 * 128 + 255) / 256, 256, 0, stream>>>(aw0, pa0, 256, 128);
    k_packw<<<(128 * 128 + 255) / 256, 256, 0, stream>>>(aw1, pa1, 128, 128);
    k_packw<<<(128 * 64 + 255) / 256, 256, 0, stream>>>(aw2, pa2, 128, 64);

    // 3) init accumulators
    size_t nm = (size_t)N * D;
    size_t ninit = nz > nm ? nz : nm;
    k_init<<<(int)((ninit + 255) / 256), 256, 0, stream>>>(accum, nz, gmax, nm);

    // 4) edge MLP + scatter (32 edges per wave, 4 waves per block)
    int etiles = (E + 31) / 32;
    k_edge<<<(etiles + 3) / 4, 128, 0, stream>>>(x0h, adj, pe0, eb0, pe1, eb1,
                                                 pe2, eb2, hbuf, gsum, gsum2,
                                                 gmax, E, etiles);

    // 5) softmax accumulation
    k_soft<<<(E + 7) / 8, 256, 0, stream>>>(hbuf, adj, gmax, gs1, gw1, gs10,
                                            gw10, gcnt, E);

    // 6) node MLP + expmap0 + proj
    int ntiles = (N + 15) / 16;
    k_node<<<(ntiles + 3) / 4, 128, 0, stream>>>(gsum, gsum2, gs1, gw1, gs10,
                                                 gw10, gcnt, pa0, ab0, pa1, ab1,
                                                 pa2, ab2, (float*)d_out, N, ntiles);
}
